// MultiHeadSelfAttention2D_6622839570522
// MI455X (gfx1250) — matrix operand order repaired and checked by reading, not compiled, not run on hardware
//
#include <hip/hip_runtime.h>

typedef __attribute__((ext_vector_type(16))) _Float16 v16h;
typedef __attribute__((ext_vector_type(8)))  _Float16 v8h;
typedef __attribute__((ext_vector_type(8)))  float    v8f;

union U16 { v16h v; v8h h[2]; };

#define B_  2
#define C_  64
#define T_  2048
#define F_  64
#define H_  4
#define HC_ 4
#define VC_ 16
#define D_  256      // HC_*F_
#define DV_ 1024     // VC_*F_

__device__ __forceinline__ v8f vzero8() {
  v8f r;
#pragma unroll
  for (int i = 0; i < 8; ++i) r[i] = 0.f;
  return r;
}

// xor-shuffle reductions within 16-lane halves via ds_swizzle (group-of-32 mode)
__device__ __forceinline__ float red_max16(float x) {
  x = fmaxf(x, __int_as_float(__builtin_amdgcn_ds_swizzle(__float_as_int(x), 0x041f)));
  x = fmaxf(x, __int_as_float(__builtin_amdgcn_ds_swizzle(__float_as_int(x), 0x081f)));
  x = fmaxf(x, __int_as_float(__builtin_amdgcn_ds_swizzle(__float_as_int(x), 0x101f)));
  x = fmaxf(x, __int_as_float(__builtin_amdgcn_ds_swizzle(__float_as_int(x), 0x201f)));
  return x;
}
__device__ __forceinline__ float red_add16(float x) {
  x += __int_as_float(__builtin_amdgcn_ds_swizzle(__float_as_int(x), 0x041f));
  x += __int_as_float(__builtin_amdgcn_ds_swizzle(__float_as_int(x), 0x081f));
  x += __int_as_float(__builtin_amdgcn_ds_swizzle(__float_as_int(x), 0x101f));
  x += __int_as_float(__builtin_amdgcn_ds_swizzle(__float_as_int(x), 0x201f));
  return x;
}

// CDNA5 async global -> LDS copy (16B per lane), tracked by ASYNCcnt
__device__ __forceinline__ void async_ld_b128(void* lds_dst, const void* gsrc) {
  unsigned dst = (unsigned)(uintptr_t)lds_dst;   // aperture: LDS addr = addr[31:0]
  unsigned long long src = (unsigned long long)(uintptr_t)gsrc;
  asm volatile("global_load_async_to_lds_b128 %0, %1, off"
               :: "v"(dst), "v"(src) : "memory");
}
__device__ __forceinline__ void wait_async_le4() {
  asm volatile("s_wait_asynccnt 0x4" ::: "memory");
}
__device__ __forceinline__ void wait_async_0() {
  asm volatile("s_wait_asynccnt 0x0" ::: "memory");
}

// ---------------- Kernel 0: x [B,C,T,F] f32 -> xT [B,T,F,C] f16 ----------------
__global__ void __launch_bounds__(256) k_xt(const float* __restrict__ x,
                                            _Float16* __restrict__ xt) {
  __shared__ _Float16 tile[64 * 65];
  const int bt = blockIdx.x;                 // (b*T + t)
  const int b = bt >> 11, t = bt & (T_ - 1);
  const float* src = x + ((size_t)b * C_ * T_ + t) * F_;   // + c*T*F + f
  for (int i = threadIdx.x; i < 64 * 64; i += 256) {
    int c = i >> 6, f = i & 63;
    tile[f * 65 + c] = (_Float16)src[(size_t)c * (T_ * F_) + f];
  }
  __syncthreads();
  _Float16* dst = xt + (size_t)bt * (F_ * C_);
  for (int i = threadIdx.x; i < 64 * 64; i += 256) {
    int f = i >> 6, c = i & 63;
    dst[i] = tile[f * 65 + c];
  }
}

// ------------- per-group PReLU + LayerNorm helper (channel dim) -------------
template <int NCH>
__device__ __forceinline__ void ln_task(const float* __restrict__ yt, int r0, int loc,
                                        const float* __restrict__ bias, float alpha,
                                        const float* __restrict__ gg,
                                        const float* __restrict__ bb,
                                        _Float16* __restrict__ dst, size_t base,
                                        size_t cstride) {
  float arr[NCH]; float s = 0.f, ss = 0.f;
#pragma unroll
  for (int c = 0; c < NCH; ++c) {
    float v = yt[(r0 + c) * 66 + loc] + bias[c];
    v = v >= 0.f ? v : alpha * v;
    arr[c] = v; s += v; ss += v * v;
  }
  float mu = s * (1.f / NCH);
  float var = ss * (1.f / NCH) - mu * mu;
  float rs = __frsqrt_rn(var + 1e-5f);
#pragma unroll
  for (int c = 0; c < NCH; ++c)
    dst[base + (size_t)c * cstride] = (_Float16)((arr[c] - mu) * rs * gg[c] + bb[c]);
}

// ---------------- Kernel A: fused QKV projection (WMMA) ----------------
__global__ void __launch_bounds__(256) k_qkv(
    const _Float16* __restrict__ xt,
    const float* __restrict__ Wq, const float* __restrict__ bq, const float* __restrict__ aq,
    const float* __restrict__ gq, const float* __restrict__ beq,
    const float* __restrict__ Wk, const float* __restrict__ bk, const float* __restrict__ ak,
    const float* __restrict__ gk, const float* __restrict__ bek,
    const float* __restrict__ Wv, const float* __restrict__ bv, const float* __restrict__ av,
    const float* __restrict__ gv, const float* __restrict__ bev,
    _Float16* __restrict__ Qh, _Float16* __restrict__ Kh, _Float16* __restrict__ Vt) {
  __shared__ __align__(32) _Float16 Wl[96 * 64];   // stacked [Wq;Wk;Wv] f16
  __shared__ float yt[96 * 66];
  const int tid = threadIdx.x;
  for (int i = tid; i < 96 * 64; i += 256) {
    int r = i >> 6; float w;
    if (r < 16)      w = Wq[i];
    else if (r < 32) w = Wk[i - 16 * 64];
    else             w = Wv[i - 32 * 64];
    Wl[i] = (_Float16)w;
  }
  __syncthreads();
  const int lane = tid & 31, wave = tid >> 5;
  const int kh = lane >> 4, ln = lane & 15;
  const _Float16* xb = xt + (size_t)blockIdx.x * (F_ * C_);  // [64 loc][64 c]
  for (int st = wave; st < 24; st += 8) {
    int rt = st >> 2, ct = st & 3;
    v8f acc = vzero8();
#pragma unroll
    for (int kc = 0; kc < 2; ++kc) {
      U16 a;
      int m = rt * 16 + ln;
      a.h[0] = *(const v8h*)&Wl[m * 64 + kc * 32 + 16 * kh];
      a.h[1] = *(const v8h*)&Wl[m * 64 + kc * 32 + 8 + 16 * kh];
      v16h bm = *(const v16h*)(xb + (ct * 16 + ln) * 64 + kc * 32 + 16 * kh);
      acc = __builtin_amdgcn_wmma_f32_16x16x32_f16(false, a.v, false, bm, (short)0, acc,
                                                   false, false);
    }
    int col = ct * 16 + ln;
#pragma unroll
    for (int v = 0; v < 8; ++v) yt[(rt * 16 + v + 8 * kh) * 66 + col] = acc[v];
  }
  __syncthreads();
  const int bt = blockIdx.x;
  const int b = bt >> 11, t = bt & (T_ - 1);
  for (int task = tid; task < 768; task += 256) {   // 12 groups x 64 locations
    int g = task >> 6, loc = task & 63, f = loc;
    int type = g >> 2, h = g & 3;
    size_t hb = (size_t)(h * B_ + b);
    if (type == 0) {
      size_t base = (hb * T_ + t) * D_ + f;
      ln_task<4>(yt, h * 4, loc, bq + h * 4, aq[h], gq + h * 4, beq + h * 4, Qh, base, 64);
    } else if (type == 1) {
      size_t base = (hb * T_ + t) * D_ + f;
      ln_task<4>(yt, 16 + h * 4, loc, bk + h * 4, ak[h], gk + h * 4, bek + h * 4, Kh, base, 64);
    } else {
      size_t base = hb * (size_t)DV_ * T_ + (size_t)f * T_ + t;  // Vt[hb][dv][t], dv=vc*64+f
      ln_task<16>(yt, 32 + h * 16, loc, bv + h * 16, av[h], gv + h * 16, bev + h * 16, Vt,
                  base, (size_t)64 * T_);
    }
  }
}

// ---------------- Kernel B: flash attention (WMMA f16, fp32 accum) ----------------
// K blocks double-buffered in LDS via CDNA5 async global->LDS copies (ASYNCcnt).
__global__ void __launch_bounds__(256) k_attn(const _Float16* __restrict__ Qh,
                                              const _Float16* __restrict__ Kh,
                                              const _Float16* __restrict__ Vt,
                                              _Float16* __restrict__ Oc) {
  __shared__ __align__(32) _Float16 Ktile[2][32 * D_];   // 2 x 16 KB
  __shared__ __align__(32) _Float16 Pbuf[8 * 16 * 32];   // per-wave P buffer, 8 KB
  const int tq = blockIdx.x, b = blockIdx.y, h = blockIdx.z;
  const int tid = threadIdx.x, lane = tid & 31, wave = tid >> 5;
  const int kh = lane >> 4, ln = lane & 15;
  const size_t hb = (size_t)(h * B_ + b);
  const _Float16* Qb = Qh + hb * T_ * D_;
  const _Float16* Kb = Kh + hb * T_ * D_;
  const _Float16* Vb = Vt + hb * (size_t)DV_ * T_;
  const int t0 = tq * 16;

  // per-lane staging coordinates (4 x 16B chunks of the 32x256 K block)
  const int skey = tid >> 5;            // rows tid>>5, +8, +16, +24
  const int soff = (tid & 31) * 8;      // 8 halfs = 16B

  U16 qa[8];                     // Q rows t0..t0+15, d=256 as 8 K=32 chunks
#pragma unroll
  for (int c = 0; c < 8; ++c) {
    const _Float16* qp = Qb + (size_t)(t0 + ln) * D_ + c * 32 + 16 * kh;
    qa[c].h[0] = *(const v8h*)qp;
    qa[c].h[1] = *(const v8h*)(qp + 8);
  }
  v8f acc[8];
#pragma unroll
  for (int j = 0; j < 8; ++j) acc[j] = vzero8();
  float M[8], L[8];
#pragma unroll
  for (int v = 0; v < 8; ++v) { M[v] = -1e30f; L[v] = 0.f; }
  _Float16* pb = Pbuf + wave * 512;
  const int dv0 = wave * 128;

  // prologue: stage K block 0 into buffer 0
#pragma unroll
  for (int i = 0; i < 4; ++i) {
    int key = skey + i * 8;
    async_ld_b128(&Ktile[0][key * D_ + soff], Kb + (size_t)key * D_ + soff);
  }

  for (int it = 0; it < T_ / 32; ++it) {
    const int s0 = it * 32;
    const int cur = it & 1;
    if (it + 1 < T_ / 32) {             // prefetch next K block into other buffer
#pragma unroll
      for (int i = 0; i < 4; ++i) {
        int key = skey + i * 8;
        async_ld_b128(&Ktile[cur ^ 1][key * D_ + soff],
                      Kb + (size_t)(s0 + 32 + key) * D_ + soff);
      }
      wait_async_le4();                 // current block's 4 copies retired
    } else {
      wait_async_0();
    }
    __syncthreads();                    // all waves' copies visible
    const _Float16* Kt = Ktile[cur];
    v8f sa = vzero8(), sb = vzero8();
#pragma unroll
    for (int c = 0; c < 8; ++c) {
      v16h b0 = *(const v16h*)&Kt[ln * D_ + c * 32 + 16 * kh];
      v16h b1 = *(const v16h*)&Kt[(16 + ln) * D_ + c * 32 + 16 * kh];
      sa = __builtin_amdgcn_wmma_f32_16x16x32_f16(false, qa[c].v, false, b0, (short)0, sa,
                                                  false, false);
      sb = __builtin_amdgcn_wmma_f32_16x16x32_f16(false, qa[c].v, false, b1, (short)0, sb,
                                                  false, false);
    }
    float r[8];
#pragma unroll
    for (int v = 0; v < 8; ++v) {
      sa[v] *= 0.0625f; sb[v] *= 0.0625f;           // 1/sqrt(256)
      float m = red_max16(fmaxf(sa[v], sb[v]));
      float nm = fmaxf(M[v], m);
      r[v] = __expf(M[v] - nm);
      M[v] = nm;
      float p0 = __expf(sa[v] - nm);
      float p1 = __expf(sb[v] - nm);
      L[v] = L[v] * r[v] + red_add16(p0 + p1);
      int row = v + 8 * kh;
      pb[row * 32 + ln]      = (_Float16)p0;
      pb[row * 32 + 16 + ln] = (_Float16)p1;
    }
#pragma unroll
    for (int j = 0; j < 8; ++j)
#pragma unroll
      for (int v = 0; v < 8; ++v) acc[j][v] *= r[v];
    U16 pa;                                          // P as A-matrix 16x32
    pa.h[0] = *(const v8h*)&pb[ln * 32 + 16 * kh];
    pa.h[1] = *(const v8h*)&pb[ln * 32 + 8 + 16 * kh];
#pragma unroll
    for (int j = 0; j < 8; ++j) {
      const _Float16* vp = Vb + (size_t)(dv0 + j * 16 + ln) * T_ + s0 + 16 * kh;
      v16h bvm = *(const v16h*)vp;
      __builtin_prefetch(vp + 32, 0, 0);             // pull next block's V rows
      acc[j] = __builtin_amdgcn_wmma_f32_16x16x32_f16(false, pa.v, false, bvm, (short)0,
                                                      acc[j], false, false);
    }
    __syncthreads();                    // done reading K buffer before it is reused
  }
#pragma unroll
  for (int j = 0; j < 8; ++j) {
    int dv = dv0 + j * 16 + ln;
    int vc = dv >> 6, f = dv & 63;
#pragma unroll
    for (int v = 0; v < 8; ++v) {
      int t = t0 + v + 8 * kh;
      float o = acc[j][v] / L[v];
      Oc[(((size_t)b * T_ + t) * F_ + f) * C_ + (h * VC_ + vc)] = (_Float16)o;
    }
  }
}

// ---------------- Kernel C: output projection + PReLU + LN + residual ----------------
__global__ void __launch_bounds__(256) k_out(const _Float16* __restrict__ Oc,
                                             const float* __restrict__ Wp,
                                             const float* __restrict__ bp,
                                             const float* __restrict__ ap,
                                             const float* __restrict__ gp,
                                             const float* __restrict__ bep,
                                             const float* __restrict__ x,
                                             float* __restrict__ out) {
  __shared__ __align__(32) _Float16 Wl[64 * 64];
  __shared__ float yt[64 * 66];
  const int tid = threadIdx.x;
  for (int i = tid; i < 64 * 64; i += 256) Wl[i] = (_Float16)Wp[i];
  __syncthreads();
  const int lane = tid & 31, wave = tid >> 5, kh = lane >> 4, ln = lane & 15;
  const _Float16* ob = Oc + (size_t)blockIdx.x * (F_ * C_);
  for (int k = 0; k < 2; ++k) {
    int st = wave + k * 8;
    int rt = st >> 2, ct = st & 3;
    v8f acc = vzero8();
#pragma unroll
    for (int kc = 0; kc < 2; ++kc) {
      U16 a;
      int m = rt * 16 + ln;
      a.h[0] = *(const v8h*)&Wl[m * 64 + kc * 32 + 16 * kh];
      a.h[1] = *(const v8h*)&Wl[m * 64 + kc * 32 + 8 + 16 * kh];
      v16h bm = *(const v16h*)(ob + (ct * 16 + ln) * 64 + kc * 32 + 16 * kh);
      acc = __builtin_amdgcn_wmma_f32_16x16x32_f16(false, a.v, false, bm, (short)0, acc,
                                                   false, false);
    }
#pragma unroll
    for (int v = 0; v < 8; ++v) yt[(rt * 16 + v + 8 * kh) * 66 + ct * 16 + ln] = acc[v];
  }
  __syncthreads();
  if (tid < 64) {
    const int f = tid, bt = blockIdx.x;
    const int b = bt >> 11, t = bt & (T_ - 1);
    const float alpha = ap[0];
    float s = 0.f, ss = 0.f;
    for (int c = 0; c < 64; ++c) {
      float v = yt[c * 66 + f] + bp[c];
      v = v >= 0.f ? v : alpha * v;
      s += v; ss += v * v;
    }
    float mu = s * (1.f / 64), var = ss * (1.f / 64) - mu * mu;
    float rs = __frsqrt_rn(var + 1e-5f);
    for (int c = 0; c < 64; ++c) {
      float v = yt[c * 66 + f] + bp[c];
      v = v >= 0.f ? v : alpha * v;
      size_t xi = (((size_t)b * C_ + c) * T_ + t) * F_ + f;
      out[xi] = (v - mu) * rs * gp[c] + bep[c] + x[xi];
    }
  }
}

extern "C" void kernel_launch(void* const* d_in, const int* in_sizes, int n_in,
                              void* d_out, int out_size, void* d_ws, size_t ws_size,
                              hipStream_t stream) {
  const float* x   = (const float*)d_in[0];
  const float* Wq  = (const float*)d_in[1];
  const float* bq  = (const float*)d_in[2];
  const float* aq  = (const float*)d_in[3];
  const float* gq  = (const float*)d_in[4];
  const float* beq = (const float*)d_in[5];
  const float* Wk  = (const float*)d_in[6];
  const float* bk  = (const float*)d_in[7];
  const float* ak  = (const float*)d_in[8];
  const float* gk  = (const float*)d_in[9];
  const float* bek = (const float*)d_in[10];
  const float* Wv  = (const float*)d_in[11];
  const float* bv  = (const float*)d_in[12];
  const float* av  = (const float*)d_in[13];
  const float* gv  = (const float*)d_in[14];
  const float* bev = (const float*)d_in[15];
  const float* Wp  = (const float*)d_in[16];
  const float* bp  = (const float*)d_in[17];
  const float* ap  = (const float*)d_in[18];
  const float* gp  = (const float*)d_in[19];
  const float* bep = (const float*)d_in[20];
  (void)in_sizes; (void)n_in; (void)out_size; (void)ws_size;

  char* ws = (char*)d_ws;
  _Float16* xT = (_Float16*)(ws);                       // 33,554,432 B
  _Float16* Qh = (_Float16*)(ws + 33554432ull);         //  8,388,608 B
  _Float16* Kh = (_Float16*)(ws + 41943040ull);         //  8,388,608 B
  _Float16* Vt = (_Float16*)(ws + 50331648ull);         // 33,554,432 B
  _Float16* Oc = (_Float16*)(ws + 83886080ull);         // 33,554,432 B

  k_xt <<<B_ * T_, 256, 0, stream>>>(x, xT);
  k_qkv<<<B_ * T_, 256, 0, stream>>>(xT, Wq, bq, aq, gq, beq, Wk, bk, ak, gk, bek,
                                     Wv, bv, av, gv, bev, Qh, Kh, Vt);
  k_attn<<<dim3(T_ / 16, B_, H_), 256, 0, stream>>>(Qh, Kh, Vt, Oc);
  k_out<<<B_ * T_, 256, 0, stream>>>(Oc, Wp, bp, ap, gp, bep, x, (float*)d_out);
}
